// EncoderLayerMoE_8289286881670
// MI455X (gfx1250) — matrix operand order repaired and checked by reading, not compiled, not run on hardware
//
#include <hip/hip_runtime.h>

#define D_MODEL 1024
#define D_FF    4096
#define NEXP    8
#define TTOK    4096   // B*S = 4*1024
#define EPS     1e-5f

typedef __attribute__((ext_vector_type(16))) __bf16          v16bf;
typedef __attribute__((ext_vector_type(8)))  float           v8f;
typedef __attribute__((ext_vector_type(16))) unsigned short  v16u;
typedef __attribute__((ext_vector_type(8)))  unsigned short  v8u;

// LDS geometry for the FFN kernel (dynamic LDS)
#define SX_STRIDE 1032            // u16 elems per X row (16B-aligned stride)
#define SH_STRIDE 136             // u16 elems per H row
#define SX_BYTES  (32 * SX_STRIDE * 2)          // 66048
#define SH_BYTES  (32 * SH_STRIDE * 2)          // 8704
#define FFN_LDS_BYTES (SX_BYTES + SH_BYTES + 32 * 4 + 32 * 4)  // 75008

// round-to-nearest-even f32 -> bf16 (stored as raw u16)
__device__ __forceinline__ unsigned short f2bf(float f) {
  unsigned int u = __float_as_uint(f);
  u += 0x7FFFu + ((u >> 16) & 1u);
  return (unsigned short)(u >> 16);
}

// A-matrix 16x32 bf16 layout: lane L (m = L&15) holds two contiguous K-runs of 8
// starting at kb=(L>>4)*8 and kb+16 (ISA 7.12.2, 16-bit A).
__device__ __forceinline__ v16bf mk_a(const unsigned short* __restrict__ row, int k, int kb) {
  v8u r1 = *(const v8u*)(row + k + kb);
  v8u r2 = *(const v8u*)(row + k + kb + 16);
  v16u v;
#pragma unroll
  for (int i = 0; i < 8; ++i) { v[i] = r1[i]; v[i + 8] = r2[i]; }
  return __builtin_bit_cast(v16bf, v);
}

// B-matrix 32x16 bf16 layout: lane L (n = L&15) holds two K-runs of 8
// starting at koff=(L>>4)*8 and koff+16. p points at [col][k0], contiguous in K.
__device__ __forceinline__ v16bf mk_b(const unsigned short* __restrict__ p) {
  v8u r1 = *(const v8u*)(p);
  v8u r2 = *(const v8u*)(p + 16);
  v16u v;
#pragma unroll
  for (int i = 0; i < 8; ++i) { v[i] = r1[i]; v[i + 8] = r2[i]; }
  return __builtin_bit_cast(v16bf, v);
}

// ---------------------------------------------------------------- init ------
__global__ void init_kernel(float* __restrict__ Y, int* __restrict__ cnt,
                            float* __restrict__ ps, size_t n) {
  size_t i = (size_t)blockIdx.x * blockDim.x + threadIdx.x;
  size_t stride = (size_t)gridDim.x * blockDim.x;
  for (size_t j = i; j < n; j += stride) Y[j] = 0.0f;
  if (i < NEXP) { cnt[i] = 0; ps[i] = 0.0f; }
}

// --------------------------------------------------------- x f32 -> bf16 ---
__global__ void cvt_x_kernel(const float* __restrict__ x,
                             unsigned short* __restrict__ xb, size_t n) {
  size_t i = (size_t)blockIdx.x * blockDim.x + threadIdx.x;
  size_t stride = (size_t)gridDim.x * blockDim.x;
  for (size_t j = i; j < n; j += stride) xb[j] = f2bf(x[j]);
}

// ---------------------------------------- transpose [e][M][N] -> bf16 [e][N][M]
__global__ void transpose_bf16_kernel(const float* __restrict__ src,
                                      unsigned short* __restrict__ dst,
                                      int M, int N) {
  __shared__ float tile[32][33];
  const int e = blockIdx.z;
  const float* s = src + (size_t)e * M * N;
  unsigned short* d = dst + (size_t)e * M * N;
  const int n0 = blockIdx.x * 32, m0 = blockIdx.y * 32;
#pragma unroll
  for (int i = 0; i < 4; ++i) {
    int m = threadIdx.y + i * 8;
    tile[m][threadIdx.x] = s[(size_t)(m0 + m) * N + n0 + threadIdx.x];
  }
  __syncthreads();
#pragma unroll
  for (int i = 0; i < 4; ++i) {
    int nn = threadIdx.y + i * 8;
    d[(size_t)(n0 + nn) * M + m0 + threadIdx.x] = f2bf(tile[threadIdx.x][nn]);
  }
}

// ---------------------------------------------------------------- gating ---
__global__ void gate_kernel(const float* __restrict__ x, const float* __restrict__ Wg,
                            int* __restrict__ lists, float* __restrict__ lwgt,
                            int* __restrict__ cnt, float* __restrict__ probsum) {
  const int t = blockIdx.x * blockDim.x + threadIdx.x;
  if (t >= TTOK) return;
  float acc[NEXP];
#pragma unroll
  for (int e = 0; e < NEXP; ++e) acc[e] = 0.0f;
  const float* xr = x + (size_t)t * D_MODEL;
  for (int d = 0; d < D_MODEL; ++d) {
    float xv = xr[d];
    const float* wr = Wg + d * NEXP;
#pragma unroll
    for (int e = 0; e < NEXP; ++e) acc[e] += xv * wr[e];
  }
  float mx = acc[0];
#pragma unroll
  for (int e = 1; e < NEXP; ++e) mx = fmaxf(mx, acc[e]);
  float p[NEXP], s = 0.0f;
#pragma unroll
  for (int e = 0; e < NEXP; ++e) { p[e] = __expf(acc[e] - mx); s += p[e]; }
  const float inv = 1.0f / s;
#pragma unroll
  for (int e = 0; e < NEXP; ++e) { p[e] *= inv; atomicAdd(&probsum[e], p[e]); }
  // top-2, first-index tie break (matches lax.top_k)
  int i1 = 0; float v1 = p[0];
#pragma unroll
  for (int e = 1; e < NEXP; ++e) if (p[e] > v1) { v1 = p[e]; i1 = e; }
  int i2 = (i1 == 0) ? 1 : 0; float v2 = p[i2];
#pragma unroll
  for (int e = 0; e < NEXP; ++e)
    if (e != i1 && e != i2 && p[e] > v2) { v2 = p[e]; i2 = e; }
  const float wsum = v1 + v2;
  const float w1 = v1 / wsum, w2 = v2 / wsum;
  int pos = atomicAdd(&cnt[i1], 1);
  lists[i1 * TTOK + pos] = t; lwgt[i1 * TTOK + pos] = w1;
  pos = atomicAdd(&cnt[i2], 1);
  lists[i2 * TTOK + pos] = t; lwgt[i2 * TTOK + pos] = w2;
}

// ---------------------------------------------------- grouped expert FFN ---
// grid: (TTOK/32, NEXP), block: 512 (16 wave32), dynamic LDS = FFN_LDS_BYTES.
// M-tile = 32 tokens: halves expert-weight L2 traffic vs M=16.
__global__ __launch_bounds__(512) void ffn_kernel(
    const unsigned short* __restrict__ xb,    // [T][D]   bf16
    const unsigned short* __restrict__ W1t,   // [E][F][D] bf16 (B of GEMM1)
    const unsigned short* __restrict__ W2t,   // [E][D][F] bf16 (B of GEMM2)
    const float* __restrict__ b1,             // [E][F]
    const float* __restrict__ b2,             // [E][D]
    const int* __restrict__ lists, const float* __restrict__ lwgt,
    const int* __restrict__ cnt, float* __restrict__ Y) {
  const int e = blockIdx.y;
  const int tile = blockIdx.x;
  const int n_tok = cnt[e];
  if (tile * 32 >= n_tok) return;

  extern __shared__ unsigned char smem[];
  unsigned short* sX   = (unsigned short*)(smem);                    // [32][SX_STRIDE]
  unsigned short* sH   = (unsigned short*)(smem + SX_BYTES);         // [32][SH_STRIDE]
  int*            sTok = (int*)  (smem + SX_BYTES + SH_BYTES);       // [32]
  float*          sW   = (float*)(smem + SX_BYTES + SH_BYTES + 128); // [32]

  const int tid = threadIdx.x;
  if (tid < 32) {
    int idx = tile * 32 + tid;
    if (idx < n_tok) { sTok[tid] = lists[e * TTOK + idx]; sW[tid] = lwgt[e * TTOK + idx]; }
    else             { sTok[tid] = lists[e * TTOK + tile * 32]; sW[tid] = 0.0f; } // pad: weight 0
  }
  __syncthreads();
  // vectorized gather: 32 rows x 128 b128-chunks
  for (int i = tid; i < 32 * 128; i += 512) {
    int m = i >> 7, c = (i & 127) * 8;
    *(v8u*)(sX + m * SX_STRIDE + c) = *(const v8u*)(xb + (size_t)sTok[m] * D_MODEL + c);
  }
  __syncthreads();

  const int wave = tid >> 5, lane = tid & 31;
  const int mh   = wave >> 3;          // which 16-token half this wave serves
  const int wsub = wave & 7;           // F-subtile (phase1) / D-tile lane (phase2)
  const int mrow = lane & 15;          // A-operand row within half
  const int kb   = (lane >> 4) * 8;    // A K-run start
  const int ncol = lane & 15;          // B-operand column
  const int koff = (lane >> 4) * 8;    // B K-run start
  const int hi   = lane >> 4;          // C-layout row-half select

  const unsigned short* aX = sX + (mh * 16 + mrow) * SX_STRIDE;
  const unsigned short* aH = sH + (mh * 16 + mrow) * SH_STRIDE;

  v8f acc2[8];
#pragma unroll
  for (int i = 0; i < 8; ++i) acc2[i] = (v8f){0.f,0.f,0.f,0.f,0.f,0.f,0.f,0.f};

  for (int fc = 0; fc < D_FF; fc += 128) {
    // ---- phase 1: wave (mh,wsub) computes h tile rows [mh*16,+16), cols [fc+16*wsub,+16)
    const int col0 = fc + wsub * 16;
    v8f a1 = (v8f){0.f,0.f,0.f,0.f,0.f,0.f,0.f,0.f};
    const unsigned short* w1p = W1t + ((size_t)e * D_FF + col0 + ncol) * D_MODEL;
    for (int k = 0; k < D_MODEL; k += 32) {
      __builtin_prefetch(w1p + k + 256, 0, 1);
      v16bf A = mk_a(aX, k, kb);
      v16bf B = mk_b(w1p + k + koff);
      a1 = __builtin_amdgcn_wmma_f32_16x16x32_bf16(false, A, false, B, (short)0, a1, false, false);
    }
    const float bias = b1[e * D_FF + col0 + ncol];
#pragma unroll
    for (int r = 0; r < 8; ++r) {                 // C layout: (M=r+8*hi, N=ncol)
      float h = a1[r] + bias;
      h = h > 0.0f ? h : 0.0f;
      sH[(mh * 16 + r + 8 * hi) * SH_STRIDE + wsub * 16 + ncol] = f2bf(h);
    }
    __syncthreads();
    // ---- phase 2: wave accumulates 8 D-tiles for its M-half over this F chunk
#pragma unroll
    for (int dti = 0; dti < 8; ++dti) {
      const int dtile = dti * 8 + wsub;
      const unsigned short* w2p =
          W2t + ((size_t)e * D_MODEL + dtile * 16 + ncol) * D_FF + fc;
      v8f c = acc2[dti];
#pragma unroll
      for (int k = 0; k < 128; k += 32) {
        v16bf A = mk_a(aH, k, kb);
        v16bf B = mk_b(w2p + k + koff);
        c = __builtin_amdgcn_wmma_f32_16x16x32_bf16(false, A, false, B, (short)0, c, false, false);
      }
      acc2[dti] = c;
    }
    __syncthreads();
  }

  // ---- epilogue: combine-weighted scatter-add
#pragma unroll
  for (int dti = 0; dti < 8; ++dti) {
    const int dtile = dti * 8 + wsub;
    const int col = dtile * 16 + ncol;
    const float b2v = b2[e * D_MODEL + col];
#pragma unroll
    for (int r = 0; r < 8; ++r) {
      const int m = mh * 16 + r + 8 * hi;
      const float val = (acc2[dti][r] + b2v) * sW[m];
      atomicAdd(&Y[(size_t)sTok[m] * D_MODEL + col], val);
    }
  }
}

// ------------------------------------------------- residual + layernorm ----
__global__ __launch_bounds__(256) void ln_kernel(const float* __restrict__ x,
                                                 const float* __restrict__ Y,
                                                 const float* __restrict__ gamma,
                                                 const float* __restrict__ beta,
                                                 float* __restrict__ out) {
  const int t = blockIdx.x, tid = threadIdx.x;
  __shared__ float red[18];
  const float* xr = x + (size_t)t * D_MODEL;
  const float* yr = Y + (size_t)t * D_MODEL;
  float v[4], s = 0.0f, ss = 0.0f;
#pragma unroll
  for (int i = 0; i < 4; ++i) {
    int d = tid + i * 256;
    float z = xr[d] + yr[d];
    v[i] = z; s += z; ss += z * z;
  }
#pragma unroll
  for (int off = 16; off > 0; off >>= 1) {
    s  += __shfl_xor(s, off, 32);
    ss += __shfl_xor(ss, off, 32);
  }
  const int wave = tid >> 5;
  if ((tid & 31) == 0) { red[wave] = s; red[8 + wave] = ss; }
  __syncthreads();
  if (tid == 0) {
    float S = 0.0f, SS = 0.0f;
#pragma unroll
    for (int w = 0; w < 8; ++w) { S += red[w]; SS += red[8 + w]; }
    float mu = S * (1.0f / D_MODEL);
    float var = SS * (1.0f / D_MODEL) - mu * mu;
    red[16] = mu;
    red[17] = rsqrtf(var + EPS);
  }
  __syncthreads();
  const float mu = red[16], rs = red[17];
#pragma unroll
  for (int i = 0; i < 4; ++i) {
    int d = tid + i * 256;
    out[(size_t)t * D_MODEL + d] = (v[i] - mu) * rs * gamma[d] + beta[d];
  }
}

// ------------------------------------------------------------- aux loss ----
__global__ void aux_kernel(const int* __restrict__ cnt, const float* __restrict__ ps,
                           float* __restrict__ out_aux) {
  if (threadIdx.x == 0 && blockIdx.x == 0) {
    float a = 0.0f;
#pragma unroll
    for (int e = 0; e < NEXP; ++e)
      a += ((float)cnt[e] / (float)TTOK) * (ps[e] / (float)TTOK);
    *out_aux = (float)NEXP * a;
  }
}

// ---------------------------------------------------------------------------
extern "C" void kernel_launch(void* const* d_in, const int* in_sizes, int n_in,
                              void* d_out, int out_size, void* d_ws, size_t ws_size,
                              hipStream_t stream) {
  const float* x     = (const float*)d_in[0];
  // d_in[1] = mask (all true; unused by the math)
  const float* Wg    = (const float*)d_in[2];
  const float* W1    = (const float*)d_in[3];
  const float* b1    = (const float*)d_in[4];
  const float* W2    = (const float*)d_in[5];
  const float* b2    = (const float*)d_in[6];
  const float* gamma = (const float*)d_in[7];
  const float* beta  = (const float*)d_in[8];
  float* out = (float*)d_out;

  char* ws = (char*)d_ws;
  const size_t ND  = (size_t)TTOK * D_MODEL;                 // 4M
  unsigned short* xb  = (unsigned short*)(ws);                       // 8 MB
  unsigned short* W1t = (unsigned short*)(ws + 8388608);             // 64 MB
  unsigned short* W2t = (unsigned short*)(ws + 75497472);            // 64 MB
  float*          Yac = (float*)         (ws + 142606336);           // 16 MB
  int*            lst = (int*)           (ws + 159383552);           // 128 KB
  float*          lwt = (float*)         (ws + 159514624);           // 128 KB
  int*            cnt = (int*)           (ws + 159645696);
  float*          psm = (float*)         (ws + 159645728);

  init_kernel<<<dim3(4096), 256, 0, stream>>>(Yac, cnt, psm, ND);
  cvt_x_kernel<<<dim3(4096), 256, 0, stream>>>(x, xb, ND);
  // W1 [E][D][F] -> W1t [E][F][D]
  transpose_bf16_kernel<<<dim3(D_FF / 32, D_MODEL / 32, NEXP), dim3(32, 8), 0, stream>>>(
      W1, W1t, D_MODEL, D_FF);
  // W2 [E][F][D] -> W2t [E][D][F]
  transpose_bf16_kernel<<<dim3(D_MODEL / 32, D_FF / 32, NEXP), dim3(32, 8), 0, stream>>>(
      W2, W2t, D_FF, D_MODEL);
  gate_kernel<<<dim3(TTOK / 256), 256, 0, stream>>>(x, Wg, lst, lwt, cnt, psm);
  ffn_kernel<<<dim3(TTOK / 32, NEXP), 512, FFN_LDS_BYTES, stream>>>(
      xb, W1t, W2t, b1, b2, lst, lwt, cnt, Yac);
  ln_kernel<<<dim3(TTOK), 256, 0, stream>>>(x, Yac, gamma, beta, out);
  aux_kernel<<<dim3(1), 32, 0, stream>>>(cnt, psm, out + ND);
  (void)in_sizes; (void)n_in; (void)out_size; (void)ws_size;
}
